// DASCO_2207613190746
// MI455X (gfx1250) — matrix operand order repaired and checked by reading, not compiled, not run on hardware
//
#include <hip/hip_runtime.h>
#include <hip/hip_bf16.h>

// ---------------------------------------------------------------------------
// Types / helpers
// ---------------------------------------------------------------------------
typedef unsigned short u16;
typedef __attribute__((ext_vector_type(16))) __bf16 v16bf;
typedef __attribute__((ext_vector_type(8)))  float  v8f;
typedef __attribute__((ext_vector_type(4)))  float  v4f;
typedef __attribute__((ext_vector_type(4)))  unsigned int v4u;
typedef int v4i_g __attribute__((vector_size(16)));  // matches builtin param

union Frag16 {
    v16bf v;
    v4u   q[2];
};

// fp32 -> bf16 (round-to-nearest-even) as raw bits.
__device__ __forceinline__ u16 f2bf(float x) {
    union { float f; unsigned int u; } c;
    c.f = x;
    unsigned int u = c.u;
    unsigned int r = u + 0x7FFFu + ((u >> 16) & 1u);
    return (u16)(r >> 16);
}

__device__ __forceinline__ v16bf load_frag(const u16* p0, const u16* p1) {
    Frag16 f;
    f.q[0] = *(const v4u*)p0;
    f.q[1] = *(const v4u*)p1;
    return f.v;
}

__device__ __forceinline__ v8f wmma_bf16(v16bf a, v16bf b, v8f c) {
    return __builtin_amdgcn_wmma_f32_16x16x32_bf16(
        false, a, false, b, (short)0, c, false, false);
}

// ---- CDNA5 async global->LDS path (guarded; sync fallback) ----
#if defined(__AMDGCN__) && defined(__has_builtin)
#if __has_builtin(__builtin_amdgcn_global_load_async_to_lds_b128) && \
    __has_builtin(__builtin_amdgcn_s_wait_asynccnt)
#define HAVE_ASYNC 1
#endif
#endif
#ifndef HAVE_ASYNC
#define HAVE_ASYNC 0
#endif

#if HAVE_ASYNC
// Param types per hipcc diagnostic: (v4i __device__(AS1)*, v4i __shared__(AS3)*,
// imm offset, imm cpol).  Generic LDS pointers carry the LDS offset in the low
// 32 bits; global pointers are identity in AS(1) -> integer casts are safe.
#define ASYNC_B128(gsrc, ldst, IMM)                                           \
    __builtin_amdgcn_global_load_async_to_lds_b128(                           \
        (__attribute__((address_space(1))) v4i_g*)(unsigned long long)(gsrc), \
        (__attribute__((address_space(3))) v4i_g*)(unsigned int)(             \
            unsigned long long)(ldst),                                        \
        (IMM), 0)
#define WAIT_ASYNC(n) __builtin_amdgcn_s_wait_asynccnt(n)
#else
#define WAIT_ASYNC(n)
#endif

// ---------------------------------------------------------------------------
// f32 -> bf16 conversion pass (vectorized)
// ---------------------------------------------------------------------------
__global__ __launch_bounds__(256)
void cvt_bf16(const float* __restrict__ s, u16* __restrict__ d, int n) {
    int i = (blockIdx.x * 256 + threadIdx.x) * 4;
    if (i + 3 < n) {
        v4f f = *(const v4f*)(s + i);
        d[i]     = f2bf(f.x);
        d[i + 1] = f2bf(f.y);
        d[i + 2] = f2bf(f.z);
        d[i + 3] = f2bf(f.w);
    } else {
        for (; i < n; ++i) d[i] = f2bf(s[i]);
    }
}

// ---------------------------------------------------------------------------
// WMMA GEMM (bf16 in, f32 accumulate): C[b] = act(A[b] @ Bmat + bias)
//   Block tile 128x64, k-step 64, double-buffered LDS; 8 waves each own a
//   32x32 sub-tile => 8 v_wmma per wave per barrier pair.  With async-to-LDS,
//   tile i+1 streams into the alternate buffer while tile i computes.
// ---------------------------------------------------------------------------
#define LSA 72  // LDS row stride in ushorts (144B: 16B-aligned, conflict-free)

template <int ACT, bool BT, bool OUTBF, bool FULL>
__global__ __launch_bounds__(256)
void gemm_bf16(const u16* __restrict__ A, long long sA,
               const u16* __restrict__ Bm, long long sB,
               const float* __restrict__ bias,
               void* __restrict__ Cv, long long sC,
               int R, int K, int N, int lda, int ldb, int ldc) {
    __shared__ u16 As[2 * 128 * LSA];
    __shared__ u16 Bs[2 * 64 * LSA];

    const int bz = blockIdx.z;
    A  += (long long)bz * sA;
    Bm += (long long)bz * sB;

    const int row0 = blockIdx.x * 128;
    const int col0 = blockIdx.y * 64;
    const int tid  = threadIdx.x;
    const int lane = tid & 31;
    const int wid  = tid >> 5;
    const int wr   = wid & 3;
    const int wc   = wid >> 2;
    const int l16  = lane & 15;
    const int half = (lane >> 4) & 1;

    v8f acc00 = {}, acc01 = {}, acc10 = {}, acc11 = {};

    auto stageA = [&](int k0, int buf) {
        const int r  = tid >> 1;
        const int cg = (tid & 1) * 32;
        const u16* src = A + (long long)(row0 + r) * lda + k0 + cg;
        u16* dst = &As[buf * 128 * LSA + r * LSA + cg];
        if (FULL) {
#if HAVE_ASYNC
            ASYNC_B128(src, dst, 0);
            ASYNC_B128(src, dst, 16);
            ASYNC_B128(src, dst, 32);
            ASYNC_B128(src, dst, 48);
#else
#pragma unroll
            for (int i = 0; i < 4; ++i)
                *(v4u*)(dst + i * 8) = *(const v4u*)(src + i * 8);
#endif
        } else {
            const bool okr = (row0 + r) < R;
#pragma unroll
            for (int i = 0; i < 4; ++i) {
                v4u v = {0u, 0u, 0u, 0u};
                if (okr && (k0 + cg + i * 8) < K) v = *(const v4u*)(src + i * 8);
                *(v4u*)(dst + i * 8) = v;
            }
        }
    };
    auto stageB = [&](int k0, int buf) {
        if (BT) {
            const int n  = tid >> 2;
            const int kg = (tid & 3) * 16;
            const u16* src = Bm + (long long)(col0 + n) * ldb + k0 + kg;
            u16* dst = &Bs[buf * 64 * LSA + n * LSA + kg];
            if (FULL) {
#if HAVE_ASYNC
                ASYNC_B128(src, dst, 0);
                ASYNC_B128(src, dst, 16);
#else
#pragma unroll
                for (int i = 0; i < 2; ++i)
                    *(v4u*)(dst + i * 8) = *(const v4u*)(src + i * 8);
#endif
            } else {
                const bool okn = (col0 + n) < N;
#pragma unroll
                for (int i = 0; i < 2; ++i) {
                    v4u v = {0u, 0u, 0u, 0u};
                    if (okn && (k0 + kg + i * 8) < K)
                        v = *(const v4u*)(src + i * 8);
                    *(v4u*)(dst + i * 8) = v;
                }
            }
        } else {
            // K x N weights: transpose-gather (sync; covered by the barrier)
            const int n  = tid & 63;
            const int kb = (tid >> 6) * 16;
            const bool okn = FULL || (col0 + n) < N;
            const u16* src = Bm + (long long)(k0 + kb) * ldb + col0 + n;
            u16* dst = &Bs[buf * 64 * LSA + n * LSA + kb];
#pragma unroll
            for (int i = 0; i < 16; ++i) {
                u16 v = 0;
                if (okn && (FULL || (k0 + kb + i) < K))
                    v = src[(long long)i * ldb];
                dst[i] = v;
            }
        }
    };

    // async instructions issued per wave per tile (in-order completion)
    enum { APT = BT ? 6 : 4 };

    const int nk = (K + 63) >> 6;
    stageA(0, 0);
    stageB(0, 0);
    for (int i = 0; i < nk; ++i) {
        const int cur = i & 1;
        if (i + 1 < nk) {
            stageA((i + 1) * 64, cur ^ 1);
            stageB((i + 1) * 64, cur ^ 1);
            if (FULL) { WAIT_ASYNC(APT); }  // tile i landed; tile i+1 in flight
        } else {
            if (FULL) { WAIT_ASYNC(0); }
        }
        __syncthreads();

        const u16* AsB = &As[cur * 128 * LSA];
        const u16* BsB = &Bs[cur * 64 * LSA];
#pragma unroll
        for (int ks = 0; ks < 2; ++ks) {
            const int kk = ks * 32;
            const int m0 = wr * 32 + l16;
            v16bf a0 = load_frag(&AsB[m0 * LSA + kk + half * 16],
                                 &AsB[m0 * LSA + kk + half * 16 + 8]);
            v16bf a1 = load_frag(&AsB[(m0 + 16) * LSA + kk + half * 16],
                                 &AsB[(m0 + 16) * LSA + kk + half * 16 + 8]);
            const int n0 = wc * 32 + l16;
            v16bf b0 = load_frag(&BsB[n0 * LSA + kk + half * 16],
                                 &BsB[n0 * LSA + kk + half * 16 + 8]);
            v16bf b1 = load_frag(&BsB[(n0 + 16) * LSA + kk + half * 16],
                                 &BsB[(n0 + 16) * LSA + kk + half * 16 + 8]);
            acc00 = wmma_bf16(a0, b0, acc00);
            acc01 = wmma_bf16(a0, b1, acc01);
            acc10 = wmma_bf16(a1, b0, acc10);
            acc11 = wmma_bf16(a1, b1, acc11);
        }
        __syncthreads();
    }

    // ---- epilogue ----
    float* Cf = (float*)Cv + (long long)bz * sC;
    u16*   Cb = (u16*)Cv + (long long)bz * sC;
#pragma unroll
    for (int ar = 0; ar < 2; ++ar) {
#pragma unroll
        for (int r = 0; r < 8; ++r) {
            const int row = row0 + wr * 32 + ar * 16 + r + 8 * half;
            if (row >= R) continue;
#pragma unroll
            for (int bc = 0; bc < 2; ++bc) {
                const int col = col0 + wc * 32 + bc * 16 + l16;
                if (col >= N) continue;
                v8f a;
                if (ar == 0) a = (bc == 0) ? acc00 : acc01;
                else         a = (bc == 0) ? acc10 : acc11;
                float v = a[r] + (bias ? bias[col] : 0.0f);
                if (ACT == 1) v = fmaxf(v, 0.0f);
                if (ACT == 3) v = (v > 0.0f) ? v : (expf(v) - 1.0f);
                if (OUTBF) Cb[(long long)row * ldc + col] = f2bf(v);
                else       Cf[(long long)row * ldc + col] = v;
            }
        }
    }
}

// ---------------------------------------------------------------------------
// Fused attention -> adjacency (bf16 Q/K in, bf16 adj out, f32 scratch accum)
// ---------------------------------------------------------------------------
#define LQS 40

__global__ __launch_bounds__(256)
void attn_adj(const u16* __restrict__ Qp, const u16* __restrict__ Kp,
              const int* __restrict__ attn_mask,
              float* __restrict__ adjF, u16* __restrict__ adjB) {
    __shared__ u16 Qs[16 * LQS];
    __shared__ u16 Ks[256 * LQS];
    __shared__ float Ssc[16 * 512];
    __shared__ float red[16 * 16];
    __shared__ float rowmax[16];
    __shared__ float rowsum[16];

    const int b    = blockIdx.y;
    const int row0 = blockIdx.x * 16;
    const int tid  = threadIdx.x;
    const int lane = tid & 31;
    const int wid  = tid >> 5;
    const int l16  = lane & 15;
    const int half = (lane >> 4) & 1;
    const float scale = rsqrtf(96.0f);

    const u16* Qb = Qp + (long long)b * 512 * 768;
    const u16* Kb = Kp + (long long)b * 512 * 768;
    const int* mk = attn_mask + b * 512;

    for (int h = 0; h < 8; ++h) {
        for (int th = 0; th < 2; ++th) {
            const int t0 = th * 256;
            v8f sc0 = {};
            v8f sc1 = {};
            for (int k0 = 0; k0 < 96; k0 += 32) {
                if (tid < 64) {
                    const int r  = tid >> 2;
                    const int cg = (tid & 3) * 8;
                    *(v4u*)&Qs[r * LQS + cg] =
                        *(const v4u*)(Qb + (long long)(row0 + r) * 768 +
                                      h * 96 + k0 + cg);
                }
                {
                    const u16* src =
                        Kb + (long long)(t0 + tid) * 768 + h * 96 + k0;
                    u16* dst = &Ks[tid * LQS];
#pragma unroll
                    for (int i = 0; i < 4; ++i)
                        *(v4u*)(dst + i * 8) = *(const v4u*)(src + i * 8);
                }
                __syncthreads();
                v16bf af = load_frag(&Qs[l16 * LQS + half * 16],
                                     &Qs[l16 * LQS + half * 16 + 8]);
                const int n0 = wid * 32 + l16;
                v16bf bf0 = load_frag(&Ks[n0 * LQS + half * 16],
                                      &Ks[n0 * LQS + half * 16 + 8]);
                const int n1 = n0 + 16;
                v16bf bf1 = load_frag(&Ks[n1 * LQS + half * 16],
                                      &Ks[n1 * LQS + half * 16 + 8]);
                sc0 = wmma_bf16(af, bf0, sc0);
                sc1 = wmma_bf16(af, bf1, sc1);
                __syncthreads();
            }
#pragma unroll
            for (int r = 0; r < 8; ++r) {
                const int rr = r + 8 * half;
                const int c0 = t0 + wid * 32 + l16;
                const int c1 = c0 + 16;
                float v0 = sc0[r] * scale;
                if (mk[c0] == 0) v0 = -10000.0f;
                float v1 = sc1[r] * scale;
                if (mk[c1] == 0) v1 = -10000.0f;
                Ssc[rr * 512 + c0] = v0;
                Ssc[rr * 512 + c1] = v1;
            }
        }
        __syncthreads();
        {
            const int row = tid >> 4;
            const int sl  = tid & 15;
            float pm = -3.0e38f;
            for (int c = sl; c < 512; c += 16)
                pm = fmaxf(pm, Ssc[row * 512 + c]);
            red[row * 16 + sl] = pm;
            __syncthreads();
            if (sl == 0) {
                float m = red[row * 16];
                for (int i = 1; i < 16; ++i) m = fmaxf(m, red[row * 16 + i]);
                rowmax[row] = m;
            }
            __syncthreads();
            const float rm = rowmax[row];
            float ps = 0.0f;
            for (int c = sl; c < 512; c += 16)
                ps += expf(Ssc[row * 512 + c] - rm);
            red[row * 16 + sl] = ps;
            __syncthreads();
            if (sl == 0) {
                float s = 0.0f;
                for (int i = 0; i < 16; ++i) s += red[row * 16 + i];
                rowsum[row] = s;
            }
            __syncthreads();
            const float inv  = 0.125f / rowsum[row];
            const int   gs   = row0 + row;
            const float atts = (float)mk[gs];
            const long long basei = ((long long)b * 512 + gs) * 512;
            for (int c = sl; c < 512; c += 16) {
                float a    = expf(Ssc[row * 512 + c] - rm) * inv;
                float prev = (h == 0) ? 0.0f : adjF[basei + c];
                float v    = prev + a;
                if (h == 7) {
                    if (c == gs) v = 1.0f;
                    adjB[basei + c] = f2bf(v * atts);
                } else {
                    adjF[basei + c] = v;
                }
            }
        }
        __syncthreads();
    }
}

// ---------------------------------------------------------------------------
// Elementwise kernels
// ---------------------------------------------------------------------------
__global__ __launch_bounds__(256)
void combine_gate(const float* __restrict__ isem, const float* __restrict__ idep,
                  u16* __restrict__ icomb, int n) {
    int i = blockIdx.x * 256 + threadIdx.x;
    if (i < n) {
        float d = idep[i];
        float g = 0.6f / (1.0f + expf(-d));
        icomb[i] = f2bf((1.0f - g) * isem[i] + g * d);
    }
}

__global__ __launch_bounds__(256)
void h_update(float* __restrict__ hl, const float* __restrict__ hout,
              u16* __restrict__ hlb, int n) {
    int i = blockIdx.x * 256 + threadIdx.x;
    if (i < n) {
        float x = hl[i];
        float g = 1.0f / (1.0f + expf(-x));
        float v = g * hout[i] + (1.0f - g) * x;
        hl[i]  = v;
        hlb[i] = f2bf(v);
    }
}

__global__ __launch_bounds__(256)
void normalize_rows(const float* __restrict__ z, u16* __restrict__ out, int dim) {
    const int w    = threadIdx.x >> 5;
    const int lane = threadIdx.x & 31;
    const long long r = (long long)blockIdx.x * 8 + w;
    const float* p = z + r * dim;
    u16* q = out + r * dim;
    float ss = 0.0f;
    for (int i = lane; i < dim; i += 32) {
        float v = p[i];
        ss += v * v;
    }
#pragma unroll
    for (int off = 16; off; off >>= 1) ss += __shfl_xor(ss, off);
    const float sc = 1.0f / fmaxf(sqrtf(ss), 1e-12f);
    for (int i = lane; i < dim; i += 32) q[i] = f2bf(p[i] * sc);
}

__global__ __launch_bounds__(256)
void transpose512(const float* __restrict__ in, float* __restrict__ out) {
    __shared__ float t[32][33];
    const long long base = (long long)blockIdx.z * 512 * 512;
    const int x0 = blockIdx.x * 32, y0 = blockIdx.y * 32;
    const int lx = threadIdx.x & 31, ly = threadIdx.x >> 5;
#pragma unroll
    for (int i = 0; i < 4; ++i)
        t[ly + i * 8][lx] =
            in[base + (long long)(y0 + ly + i * 8) * 512 + x0 + lx];
    __syncthreads();
#pragma unroll
    for (int i = 0; i < 4; ++i)
        out[base + (long long)(x0 + ly + i * 8) * 512 + y0 + lx] =
            t[lx][ly + i * 8];
}

// ---------------------------------------------------------------------------
// Contrastive scope loss per row (one wave per (b,s) row); row-major reads.
// ---------------------------------------------------------------------------
__global__ __launch_bounds__(256)
void loss_rows(const float* __restrict__ base, const float* __restrict__ baseT,
               const float* __restrict__ refl1, const float* __restrict__ refl2,
               const int* __restrict__ s_mask, const int* __restrict__ a_mask,
               float* __restrict__ loss) {
    const int b    = blockIdx.y;
    const int w    = threadIdx.x >> 5;
    const int lane = threadIdx.x & 31;
    const int s    = blockIdx.x * 8 + w;
    const float it = 1.0f / 0.07f;

    const long long rb = ((long long)b * 512 + s) * 512;
    const float am_s = (float)a_mask[b * 512 + s];
    const float sm_s = (float)s_mask[b * 512 + s];
    const float base_ss = base[rb + s];
    const float r1_ss   = refl1[rb + s];
    const float r2_ss   = refl2[rb + s];
    const float wdiag   = am_s * sm_s * base_ss;

    float s_ar1 = 0, s_w1 = 0, s_mr1 = 0, s_mb1 = 0;
    float s_ar2 = 0, s_w2 = 0, s_mr2 = 0, s_mb2 = 0;
    for (int t = lane; t < 512; t += 32) {
        float sm_t = (float)s_mask[b * 512 + t];
        float bst  = base[rb + t];
        float bts  = baseT[rb + t];
        float r1   = refl1[rb + t];
        float r2   = refl2[rb + t];
        s_ar1 += expf(am_s * sm_t * r1 * it);
        s_w1  += expf(am_s * sm_t * bst * wdiag * it);
        s_mr1 += expf(am_s * r1 * it);
        s_mb1 += expf(am_s * bst * it);
        s_ar2 += expf(am_s * sm_t * r2 * it);
        s_w2  += expf(am_s * sm_t * bts * wdiag * it);
        s_mr2 += expf(am_s * r2 * it);
        s_mb2 += expf(am_s * bts * it);
    }
#pragma unroll
    for (int off = 16; off; off >>= 1) {
        s_ar1 += __shfl_xor(s_ar1, off);
        s_w1  += __shfl_xor(s_w1, off);
        s_mr1 += __shfl_xor(s_mr1, off);
        s_mb1 += __shfl_xor(s_mb1, off);
        s_ar2 += __shfl_xor(s_ar2, off);
        s_w2  += __shfl_xor(s_w2, off);
        s_mr2 += __shfl_xor(s_mr2, off);
        s_mb2 += __shfl_xor(s_mb2, off);
    }
    if (lane == 0) {
        const float dterm = expf(wdiag * it);
        const float wwd   = expf(wdiag * wdiag * it);
        float pos1 = dterm + (s_ar1 - expf(am_s * sm_s * r1_ss * it)) + (s_w1 - wwd);
        float all1 = s_mr1 + s_mb1 - expf(am_s * r1_ss * it);
        float l1   = -logf(pos1 / all1);
        float pos2 = dterm + (s_ar2 - expf(am_s * sm_s * r2_ss * it)) + (s_w2 - wwd);
        float all2 = s_mr2 + s_mb2 - expf(am_s * r2_ss * it);
        float l2   = -logf(pos2 / all2);
        loss[b * 512 + s] = 0.5f * (l1 + l2);
    }
}

__global__ __launch_bounds__(256)
void final_mean(const float* __restrict__ loss, float* __restrict__ out) {
    __shared__ float sd[256];
    float s = 0.0f;
    for (int i = threadIdx.x; i < 32 * 512; i += 256) s += loss[i];
    sd[threadIdx.x] = s;
    __syncthreads();
    for (int off = 128; off; off >>= 1) {
        if (threadIdx.x < off) sd[threadIdx.x] += sd[threadIdx.x + off];
        __syncthreads();
    }
    if (threadIdx.x == 0) out[0] = sd[0] / (32.0f * 512.0f);
}

// ---------------------------------------------------------------------------
// Host orchestration
// ---------------------------------------------------------------------------
extern "C" void kernel_launch(void* const* d_in, const int* in_sizes, int n_in,
                              void* d_out, int out_size, void* d_ws, size_t ws_size,
                              hipStream_t stream) {
    (void)in_sizes; (void)n_in; (void)out_size; (void)ws_size;

    const int Bc = 32, Sc = 512, Dc = 768, Mc = 384;
    const int Rall = Bc * Sc;
    const long long SS = (long long)Sc * Sc;
    const long long SD = (long long)Sc * Dc;
    const long long SM = (long long)Sc * Mc;

    const float* X     = (const float*)d_in[0];
    const float* adjm  = (const float*)d_in[1];
    const int* amask   = (const int*)d_in[2];
    const int* smaskv  = (const int*)d_in[3];
    const int* amaskv  = (const int*)d_in[4];
    const float* Wq = (const float*)d_in[5];
    const float* bq = (const float*)d_in[6];
    const float* Wk = (const float*)d_in[7];
    const float* bk = (const float*)d_in[8];
    const float* semW0 = (const float*)d_in[9];
    const float* semb0 = (const float*)d_in[10];
    const float* semW1 = (const float*)d_in[11];
    const float* semb1 = (const float*)d_in[12];
    const float* depW0 = (const float*)d_in[13];
    const float* depb0 = (const float*)d_in[14];
    const float* depW1 = (const float*)d_in[15];
    const float* depb1 = (const float*)d_in[16];
    const float* fc1W = (const float*)d_in[17];
    const float* fc1b = (const float*)d_in[18];
    const float* fc2W = (const float*)d_in[19];
    const float* fc2b = (const float*)d_in[20];
    const float* fc3W = (const float*)d_in[21];
    const float* fc3b = (const float*)d_in[22];
    const float* fc4W = (const float*)d_in[23];
    const float* fc4b = (const float*)d_in[24];

    // ---- f32 workspace pool ----
    float* ws = (float*)d_ws;
    size_t off = 0;
    float* adjF  = ws + off; off += (size_t)Bc * SS;
    float* Isem  = ws + off; off += (size_t)Bc * SM;
    float* Idep  = ws + off; off += (size_t)Bc * SM;
    float* Hout  = ws + off; off += (size_t)Bc * SM;
    float* Hbuf  = ws + off; off += (size_t)Bc * SM;
    float* h1    = ws + off; off += (size_t)Bc * SM;
    float* h2    = ws + off; off += (size_t)Bc * SM;
    float* refl1 = ws + off; off += (size_t)Bc * SS;
    float* refl2 = ws + off; off += (size_t)Bc * SS;
    float* baseT = ws + off; off += (size_t)Bc * SS;
    float* lrow  = ws + off; off += (size_t)Rall;
    float* baseM = adjF;  // alias: adjF dead after attention

    // ---- bf16 workspace pool ----
    u16* wb = (u16*)(ws + off);
    size_t ob = 0;
    u16* Xb     = wb + ob; ob += (size_t)Bc * SD;
    u16* adjmB  = wb + ob; ob += (size_t)Bc * SS;
    u16* QpB    = wb + ob; ob += (size_t)Bc * SD;
    u16* KpB    = wb + ob; ob += (size_t)Bc * SD;
    u16* adjB   = wb + ob; ob += (size_t)Bc * SS;
    u16* IcomB  = wb + ob; ob += (size_t)Bc * SM;
    u16* HbufB  = wb + ob; ob += (size_t)Bc * SM;
    u16* IsemB  = wb + ob; ob += (size_t)Bc * SM;
    u16* t32B   = wb + ob; ob += (size_t)Rall * 32;
    u16* n1B    = wb + ob; ob += (size_t)Bc * SM;
    u16* n2B    = wb + ob; ob += (size_t)Bc * SM;
    u16* WqB    = wb + ob; ob += (size_t)Dc * Dc;
    u16* WkB    = wb + ob; ob += (size_t)Dc * Dc;
    u16* semW0B = wb + ob; ob += (size_t)Dc * Mc;
    u16* depW0B = wb + ob; ob += (size_t)Dc * Mc;
    u16* fc4WB  = wb + ob; ob += (size_t)Dc * Mc;
    u16* semW1B = wb + ob; ob += (size_t)Mc * Mc;
    u16* depW1B = wb + ob; ob += (size_t)Mc * Mc;
    u16* fc3WB  = wb + ob; ob += (size_t)Mc * Mc;
    u16* fc1WB  = wb + ob; ob += (size_t)Mc * 32;
    u16* fc2WB  = wb + ob; ob += (size_t)32 * Mc;
    u16* AHsB = QpB;
    u16* AHdB = KpB;

#define CVT(src, dst, n) \
    cvt_bf16<<<(((n) / 4) + 255) / 256, 256, 0, stream>>>(src, dst, n)
#define GEMM(ACT, BT, OUTBF, FULL, Ap, sA, Bp, sB, biasp, Cp, sC, R, K, N, lda, ldb, ldc, nb) \
    gemm_bf16<ACT, BT, OUTBF, FULL><<<dim3(((R) + 127) / 128, ((N) + 63) / 64, (nb)),         \
                                      256, 0, stream>>>(                                      \
        Ap, sA, Bp, sB, biasp, (void*)(Cp), sC, R, K, N, lda, ldb, ldc)

    // ---- one-time bf16 conversions ----
    CVT(X, Xb, Rall * Dc);
    CVT(adjm, adjmB, (int)(Bc * SS));
    CVT(Wq, WqB, Dc * Dc);
    CVT(Wk, WkB, Dc * Dc);
    CVT(semW0, semW0B, Dc * Mc);
    CVT(depW0, depW0B, Dc * Mc);
    CVT(fc4W, fc4WB, Dc * Mc);
    CVT(semW1, semW1B, Mc * Mc);
    CVT(depW1, depW1B, Mc * Mc);
    CVT(fc3W, fc3WB, Mc * Mc);
    CVT(fc1W, fc1WB, Mc * 32);
    CVT(fc2W, fc2WB, 32 * Mc);

    // ---- Q/K projections (bf16 out) ----
    GEMM(0, false, true, true, Xb, 0LL, WqB, 0LL, bq, QpB, 0LL, Rall, Dc, Dc, Dc, Dc, Dc, 1);
    GEMM(0, false, true, true, Xb, 0LL, WkB, 0LL, bk, KpB, 0LL, Rall, Dc, Dc, Dc, Dc, Dc, 1);

    // ---- attention -> adj (bf16) ----
    attn_adj<<<dim3(Sc / 16, Bc), 256, 0, stream>>>(QpB, KpB, amask, adjF, adjB);

    const int nM = Rall * Mc;
    const int ew = (nM + 255) / 256;

    // ---- GCN layer 0 ----
    GEMM(0, false, true, true, adjB, SS, Xb, SD, nullptr, AHsB, SD, Sc, Sc, Dc, Sc, Dc, Dc, Bc);
    GEMM(0, false, false, true, AHsB, 0LL, semW0B, 0LL, semb0, Isem, 0LL, Rall, Dc, Mc, Dc, Mc, Mc, 1);
    GEMM(0, false, true, true, adjmB, SS, Xb, SD, nullptr, AHdB, SD, Sc, Sc, Dc, Sc, Dc, Dc, Bc);
    GEMM(0, false, false, true, AHdB, 0LL, depW0B, 0LL, depb0, Idep, 0LL, Rall, Dc, Mc, Dc, Mc, Mc, 1);
    combine_gate<<<ew, 256, 0, stream>>>(Isem, Idep, IcomB, nM);
    GEMM(1, false, false, true, IcomB, 0LL, fc3WB, 0LL, fc3b, Hout, 0LL, Rall, Mc, Mc, Mc, Mc, Mc, 1);
    GEMM(0, false, false, true, Xb, 0LL, fc4WB, 0LL, fc4b, Hbuf, 0LL, Rall, Dc, Mc, Dc, Mc, Mc, 1);
    h_update<<<ew, 256, 0, stream>>>(Hbuf, Hout, HbufB, nM);

    // ---- GCN layer 1 ----
    GEMM(0, false, true, true, adjB, SS, HbufB, SM, nullptr, AHsB, SM, Sc, Sc, Mc, Sc, Mc, Mc, Bc);
    GEMM(0, false, false, true, AHsB, 0LL, semW1B, 0LL, semb1, Isem, 0LL, Rall, Mc, Mc, Mc, Mc, Mc, 1);
    GEMM(0, false, true, true, adjmB, SS, HbufB, SM, nullptr, AHdB, SM, Sc, Sc, Mc, Sc, Mc, Mc, Bc);
    GEMM(0, false, false, true, AHdB, 0LL, depW1B, 0LL, depb1, Idep, 0LL, Rall, Mc, Mc, Mc, Mc, Mc, 1);
    combine_gate<<<ew, 256, 0, stream>>>(Isem, Idep, IcomB, nM);
    GEMM(1, false, false, true, IcomB, 0LL, fc3WB, 0LL, fc3b, Hout, 0LL, Rall, Mc, Mc, Mc, Mc, Mc, 1);
    h_update<<<ew, 256, 0, stream>>>(Hbuf, Hout, HbufB, nM);

    // ---- projection heads ----
    GEMM(3, false, true, false, HbufB, 0LL, fc1WB, 0LL, fc1b, t32B, 0LL, Rall, Mc, 32, Mc, 32, 32, 1);
    GEMM(0, false, false, false, t32B, 0LL, fc2WB, 0LL, fc2b, h1, 0LL, Rall, 32, Mc, 32, Mc, Mc, 1);
    CVT(Isem, IsemB, nM);
    GEMM(3, false, true, false, IsemB, 0LL, fc1WB, 0LL, fc1b, t32B, 0LL, Rall, Mc, 32, Mc, 32, 32, 1);
    GEMM(0, false, false, false, t32B, 0LL, fc2WB, 0LL, fc2b, h2, 0LL, Rall, 32, Mc, 32, Mc, Mc, 1);

    // ---- normalize rows -> bf16 ----
    normalize_rows<<<Rall / 8, 256, 0, stream>>>(h1, n1B, Mc);
    normalize_rows<<<Rall / 8, 256, 0, stream>>>(h2, n2B, Mc);

    // ---- Gram matrices (A @ B^T) ----
    GEMM(0, true, false, true, n1B, SM, n2B, SM, nullptr, baseM, SS, Sc, Mc, Sc, Mc, Mc, Sc, Bc);
    GEMM(0, true, false, true, n1B, SM, n1B, SM, nullptr, refl1, SS, Sc, Mc, Sc, Mc, Mc, Sc, Bc);
    GEMM(0, true, false, true, n2B, SM, n2B, SM, nullptr, refl2, SS, Sc, Mc, Sc, Mc, Mc, Sc, Bc);
    transpose512<<<dim3(16, 16, Bc), 256, 0, stream>>>(baseM, baseT);

    // ---- loss + mean ----
    loss_rows<<<dim3(Sc / 8, Bc), 256, 0, stream>>>(baseM, baseT, refl1, refl2,
                                                    smaskv, amaskv, lrow);
    final_mean<<<1, 256, 0, stream>>>(lrow, (float*)d_out);

#undef CVT
#undef GEMM
}
